// MambaInMamba_33509334844072
// MI455X (gfx1250) — matrix operand order repaired and checked by reading, not compiled, not run on hardware
//
#include <hip/hip_runtime.h>

// ---------------------------------------------------------------------------
// Problem constants (from reference)
// ---------------------------------------------------------------------------
constexpr int B_   = 2;
constexpr int L_   = 1024;
constexpr int DM_  = 1024;
constexpr int N_   = 16;
constexpr int DI_  = 2048;
constexpr int DTR_ = 64;
constexpr int XD_  = DTR_ + 2 * N_;   // 96, x_dbl inner dim
constexpr int MDI_ = 64;
constexpr int MN_  = 4;
constexpr int MXD_ = 12;              // MDTR + 2*MN

// ---------------------------------------------------------------------------
// CDNA5 WMMA types
// ---------------------------------------------------------------------------
typedef __attribute__((ext_vector_type(16))) __bf16          v16bf;
typedef __attribute__((ext_vector_type(8)))  float           v8f;
typedef __attribute__((ext_vector_type(8)))  unsigned short  ushort8;
typedef __attribute__((ext_vector_type(16))) unsigned short  ushort16;

union FragA { ushort8  h[2]; v16bf v; };
union FragB { ushort16 u;    v16bf v; };

// ---------------------------------------------------------------------------
// Math helpers
// ---------------------------------------------------------------------------
__device__ __forceinline__ float sigmoidf_(float x) { return 1.f / (1.f + __expf(-x)); }
__device__ __forceinline__ float siluf_(float x)    { return x * sigmoidf_(x); }
__device__ __forceinline__ float softplusf_(float x){ return (x > 20.f) ? x : log1pf(__expf(x)); }
__device__ __forceinline__ unsigned short f2bf(float f) {
  unsigned int u = __float_as_uint(f);
  u += 0x7fffu + ((u >> 16) & 1u);     // round-to-nearest-even
  return (unsigned short)(u >> 16);
}

// ---------------------------------------------------------------------------
// fp32 -> bf16 convert (grid-stride)
// ---------------------------------------------------------------------------
__global__ void cvt_bf16(const float* __restrict__ in, unsigned short* __restrict__ out, int n) {
  int i = blockIdx.x * blockDim.x + threadIdx.x;
  int stride = gridDim.x * blockDim.x;
  for (; i < n; i += stride) out[i] = f2bf(in[i]);
}

// ---------------------------------------------------------------------------
// WMMA bf16 GEMM: C[M,N] = A[M,K] * Bw[N,K]^T   (both row-major, K contiguous)
// Fully templated dims: B-tile offsets (j*16*K elements) are compile-time and
// fold into the global_load_b128 24-bit immediate-offset field -> one base
// pointer per matrix instead of NT pointer pairs, minimal VGPR pressure so
// both double-buffer stages stay in registers.
// Each wave computes a 16 x (16*NT) strip: one A fragment feeds NT WMMAs.
// MODE 0: C row-major (m*N + n)
// MODE 1: channel-major store: m=(b,l), l=m%Ldim -> C[(b*N + n)*Ldim + l]
// ---------------------------------------------------------------------------
#define WPB 4
template <int MODE, int NT, int Nn, int K, int Ldim>
__global__ void __launch_bounds__(WPB * 32, 1) __attribute__((amdgpu_waves_per_eu(1)))
gemm_bf16_wmma(const unsigned short* __restrict__ A,
               const unsigned short* __restrict__ Bw,
               float* __restrict__ C) {
  const int lane = threadIdx.x & 31;
  const int wave = threadIdx.x >> 5;
  constexpr int ngroups = Nn / (16 * NT);
  const int tile = blockIdx.x * WPB + wave;
  const int mt = tile / ngroups;
  const int ng = tile - mt * ngroups;

  // A fragment (16-bit A 16x32): lane m = lane&15; chunks at k0+(lane>>4)*16, +8
  const int arow = (mt << 4) + (lane & 15);
  const int ka = (lane >> 4) << 4;              // 0 or 16
  const int kb = (lane >> 4) << 4;              // 0 or 16

  const unsigned short* Ap = A + (size_t)arow * K;
  // Single B base; per-tile j*16*K is a compile-time immediate offset.
  const unsigned short* Bp = Bw + (size_t)(ng * NT * 16 + (lane & 15)) * K;

  FragA fa[2];
  FragB fb[2][NT];

  // prologue: stage 0
  fa[0].h[0] = *(const ushort8*)(Ap + ka);
  fa[0].h[1] = *(const ushort8*)(Ap + 8 + ka);
#pragma unroll
  for (int j = 0; j < NT; ++j) fb[0][j].u = *(const ushort16*)(Bp + j * 16 * K + kb);

  v8f acc[NT];
#pragma unroll
  for (int j = 0; j < NT; ++j) acc[j] = (v8f){0.f, 0.f, 0.f, 0.f, 0.f, 0.f, 0.f, 0.f};

#pragma unroll 2
  for (int k0 = 0; k0 < K; k0 += 32) {
    const int cur = (k0 >> 5) & 1;
    const int nxt = cur ^ 1;
    const int k1 = k0 + 32;
    if (k1 < K) {   // issue next stage's loads before consuming current stage
      fa[nxt].h[0] = *(const ushort8*)(Ap + k1 + ka);
      fa[nxt].h[1] = *(const ushort8*)(Ap + k1 + 8 + ka);
#pragma unroll
      for (int j = 0; j < NT; ++j)
        fb[nxt][j].u = *(const ushort16*)(Bp + j * 16 * K + k1 + kb);
      if (k1 + 32 < K) {                         // gfx1250 global_prefetch_b8
        __builtin_prefetch(Ap + k1 + 32, 0, 1);
        __builtin_prefetch(Bp + k1 + 32, 0, 1);
      }
    }
#pragma unroll
    for (int j = 0; j < NT; ++j)
      acc[j] = __builtin_amdgcn_wmma_f32_16x16x32_bf16(false, fa[cur].v, false, fb[cur][j].v,
                                                       (short)0, acc[j], false, false);
  }

  // C/D layout: VGPR r -> M = r + (lane>>4)*8, N = lane&15
  const int mbase = (mt << 4) + ((lane >> 4) << 3);
#pragma unroll
  for (int j = 0; j < NT; ++j) {
    const int n = ng * NT * 16 + j * 16 + (lane & 15);
#pragma unroll
    for (int r = 0; r < 8; ++r) {
      const int m = mbase + r;
      const float v = acc[j][r];
      if (MODE == 0) {
        C[(size_t)m * Nn + n] = v;
      } else {
        const int bb = m / Ldim;
        const int l  = m - bb * Ldim;
        C[((size_t)bb * Nn + n) * Ldim + l] = v;
      }
    }
  }
}

// ---------------------------------------------------------------------------
// Fused causal depthwise conv (K=4) + SiLU over (B, DI, L).
// Writes fp32 (b,d,l) for the scan and bf16 (b,l,d) transpose for GEMM2's A.
// One thread per (b,d) channel, sequential over L with register window.
// ---------------------------------------------------------------------------
__global__ void __launch_bounds__(256)
conv4_silu(const float* __restrict__ xz, const float* __restrict__ cw,
           const float* __restrict__ cb, float* __restrict__ xc,
           unsigned short* __restrict__ xT) {
  const int t = blockIdx.x * blockDim.x + threadIdx.x;   // [0, B*DI)
  const int b = t / DI_;
  const int d = t - b * DI_;
  const float* src = xz + ((size_t)b * (2 * DI_) + d) * L_;   // x half of xz
  float* dst = xc + ((size_t)b * DI_ + d) * L_;
  unsigned short* dT = xT + (size_t)b * L_ * DI_ + d;
  const float w0 = cw[d * 4 + 0], w1 = cw[d * 4 + 1];
  const float w2 = cw[d * 4 + 2], w3 = cw[d * 4 + 3];
  const float bb = cb[d];
  float h0 = 0.f, h1 = 0.f, h2 = 0.f;
  for (int l0 = 0; l0 < L_; l0 += 4) {
    float4 x4 = *(const float4*)(src + l0);
    float xv[4] = {x4.x, x4.y, x4.z, x4.w};
    float o[4];
#pragma unroll
    for (int i = 0; i < 4; ++i) {
      const float cur = xv[i];
      const float y = w0 * h0 + w1 * h1 + w2 * h2 + w3 * cur + bb;
      h0 = h1; h1 = h2; h2 = cur;
      o[i] = siluf_(y);
      dT[(size_t)(l0 + i) * DI_] = f2bf(o[i]);
    }
    *(float4*)(dst + l0) = make_float4(o[0], o[1], o[2], o[3]);
  }
}

// ---------------------------------------------------------------------------
// Fully-fused inner Mamba block. One block per batch, 64 threads (one per
// inner channel). All weights preloaded into LDS (~54 KB of the 320 KB WGP
// pool). Sequential over L with register conv window + register scan state.
// Emits bf16 dt_m in (b,l,r) layout for the delta GEMM.
// ---------------------------------------------------------------------------
__global__ void __launch_bounds__(64)
inner_mamba(const float* __restrict__ xdbl,
            const float* __restrict__ minw, const float* __restrict__ mcw,
            const float* __restrict__ mcb,  const float* __restrict__ mxp,
            const float* __restrict__ mdtw, const float* __restrict__ mdtb,
            const float* __restrict__ mAlog,const float* __restrict__ mD,
            const float* __restrict__ moutw, unsigned short* __restrict__ dtm) {
  __shared__ float s_minw[2 * MDI_ * MDI_];   // 128x64
  __shared__ float s_moutw[DTR_ * MDI_];      // 64x64
  __shared__ float s_mxp[MXD_ * MDI_];        // 12x64
  __shared__ float s_mdtw[MDI_ * 4];
  __shared__ float s_row[MDI_], s_xs[MDI_], s_small[MXD_], s_y[MDI_];

  const int tid = threadIdx.x;
  const int b = blockIdx.x;
  for (int i = tid; i < 2 * MDI_ * MDI_; i += 64) s_minw[i] = minw[i];
  for (int i = tid; i < DTR_ * MDI_;     i += 64) s_moutw[i] = moutw[i];
  for (int i = tid; i < MXD_ * MDI_;     i += 64) s_mxp[i] = mxp[i];
  for (int i = tid; i < MDI_ * 4;        i += 64) s_mdtw[i] = mdtw[i];

  float aA[MN_], st[MN_];
#pragma unroll
  for (int n = 0; n < MN_; ++n) { aA[n] = -__expf(mAlog[tid * MN_ + n]); st[n] = 0.f; }
  const float w0 = mcw[tid * 2], w1 = mcw[tid * 2 + 1], cbv = mcb[tid];
  const float dtb = mdtb[tid], dD = mD[tid];
  float prevx = 0.f;
  __syncthreads();

  for (int l = 0; l < L_; ++l) {
    s_row[tid] = xdbl[((size_t)b * L_ + l) * XD_ + tid];   // dt_in = x_dbl[..., :64]
    __syncthreads();
    float xa = 0.f, za = 0.f;
#pragma unroll 8
    for (int dd = 0; dd < MDI_; ++dd) {
      const float r = s_row[dd];
      xa += r * s_minw[tid * MDI_ + dd];
      za += r * s_minw[(MDI_ + tid) * MDI_ + dd];
    }
    const float xs = siluf_(w0 * prevx + w1 * xa + cbv);   // causal conv K=2 + silu
    prevx = xa;
    s_xs[tid] = xs;
    __syncthreads();
    if (tid < MXD_) {
      float a = 0.f;
      for (int dd = 0; dd < MDI_; ++dd) a += s_xs[dd] * s_mxp[tid * MDI_ + dd];
      s_small[tid] = a;
    }
    __syncthreads();
    const float dt = softplusf_(s_mdtw[tid * 4 + 0] * s_small[0] +
                                s_mdtw[tid * 4 + 1] * s_small[1] +
                                s_mdtw[tid * 4 + 2] * s_small[2] +
                                s_mdtw[tid * 4 + 3] * s_small[3] + dtb);
    float acc = 0.f;
#pragma unroll
    for (int n = 0; n < MN_; ++n) {
      st[n] = __expf(dt * aA[n]) * st[n] + dt * s_small[4 + n] * xs;
      acc += st[n] * s_small[8 + n];
    }
    const float y = (acc + xs * dD) * siluf_(za);
    s_y[tid] = y;
    __syncthreads();
    float o = 0.f;
#pragma unroll 8
    for (int dd = 0; dd < MDI_; ++dd) o += s_y[dd] * s_moutw[tid * MDI_ + dd];
    dtm[((size_t)b * L_ + l) * DTR_ + tid] = f2bf(o);
    __syncthreads();
  }
}

// ---------------------------------------------------------------------------
// Outer selective scan. 16 persistent blocks of 256 threads; each thread owns
// one (b,d) channel with N=16 state in registers. B/C rows staged in LDS per
// 4-step chunk; delta/u/z read as float4. Emits bf16 y^T (b,l,d) for GEMM4.
// ---------------------------------------------------------------------------
__global__ void __launch_bounds__(256)
scan_outer(const float* __restrict__ delta, const float* __restrict__ xc,
           const float* __restrict__ xz,    const float* __restrict__ xdbl,
           const float* __restrict__ Alog,  const float* __restrict__ Dv,
           const float* __restrict__ dtbv,  unsigned short* __restrict__ yT) {
  const int tid = threadIdx.x;
  const int b = blockIdx.x >> 3;
  const int d = ((blockIdx.x & 7) << 8) + tid;
  __shared__ float s_bc[4][2 * N_];

  float aA[N_], st[N_];
#pragma unroll
  for (int n = 0; n < N_; ++n) { aA[n] = -__expf(Alog[d * N_ + n]); st[n] = 0.f; }
  const float Dd = Dv[d], dtb = dtbv[d];
  const float* dl = delta + ((size_t)b * DI_ + d) * L_;
  const float* up = xc    + ((size_t)b * DI_ + d) * L_;
  const float* zp = xz    + ((size_t)b * 2 * DI_ + DI_ + d) * L_;   // z half
  unsigned short* yo = yT + (size_t)b * L_ * DI_ + d;

  for (int l0 = 0; l0 < L_; l0 += 4) {
    if (tid < 128) {
      const int il = tid >> 5, c = tid & 31;
      s_bc[il][c] = xdbl[((size_t)b * L_ + l0 + il) * XD_ + DTR_ + c];
    }
    __syncthreads();
    float4 d4 = *(const float4*)(dl + l0);
    float4 u4 = *(const float4*)(up + l0);
    float4 z4 = *(const float4*)(zp + l0);
    float da[4] = {d4.x, d4.y, d4.z, d4.w};
    float ua[4] = {u4.x, u4.y, u4.z, u4.w};
    float za[4] = {z4.x, z4.y, z4.z, z4.w};
#pragma unroll
    for (int il = 0; il < 4; ++il) {
      const float dt = softplusf_(da[il] + dtb);
      const float u = ua[il];
      float acc = 0.f;
#pragma unroll
      for (int n = 0; n < N_; ++n) {
        st[n] = __expf(dt * aA[n]) * st[n] + dt * s_bc[il][n] * u;
        acc += st[n] * s_bc[il][N_ + n];
      }
      const float y = (acc + u * Dd) * siluf_(za[il]);
      yo[(size_t)(l0 + il) * DI_] = f2bf(y);
    }
    __syncthreads();
  }
}

// ---------------------------------------------------------------------------
// Host launcher
// ---------------------------------------------------------------------------
extern "C" void kernel_launch(void* const* d_in, const int* in_sizes, int n_in,
                              void* d_out, int out_size, void* d_ws, size_t ws_size,
                              hipStream_t stream) {
  (void)in_sizes; (void)n_in; (void)out_size; (void)ws_size;
  const float* hidden   = (const float*)d_in[0];
  const float* inprojw  = (const float*)d_in[1];
  const float* convw    = (const float*)d_in[2];
  const float* convb    = (const float*)d_in[3];
  const float* xprojw   = (const float*)d_in[4];
  const float* Alog     = (const float*)d_in[5];
  const float* Dvec     = (const float*)d_in[6];
  const float* dtoutw   = (const float*)d_in[7];
  const float* dtoutb   = (const float*)d_in[8];
  const float* outprojw = (const float*)d_in[9];
  const float* m_inw    = (const float*)d_in[10];
  const float* m_convw  = (const float*)d_in[11];
  const float* m_convb  = (const float*)d_in[12];
  const float* m_xprojw = (const float*)d_in[13];
  const float* m_dtw    = (const float*)d_in[14];
  const float* m_dtb    = (const float*)d_in[15];
  const float* m_Alog   = (const float*)d_in[16];
  const float* m_Dvec   = (const float*)d_in[17];
  const float* m_outw   = (const float*)d_in[18];
  float* out = (float*)d_out;

  // Workspace carving (aligned to 256B)
  char* w = (char*)d_ws;
  auto take = [&](size_t bytes) -> char* {
    char* p = w;
    w += (bytes + 255) & ~(size_t)255;
    return p;
  };
  unsigned short* h_bf     = (unsigned short*)take((size_t)B_ * L_ * DM_ * 2);
  unsigned short* inp_bf   = (unsigned short*)take((size_t)2 * DI_ * DM_ * 2);
  float*          xzf      = (float*)take((size_t)B_ * 2 * DI_ * L_ * 4);
  float*          xcf      = (float*)take((size_t)B_ * DI_ * L_ * 4);
  unsigned short* xT_bf    = (unsigned short*)take((size_t)B_ * L_ * DI_ * 2);
  unsigned short* xp_bf    = (unsigned short*)take((size_t)XD_ * DI_ * 2);
  float*          xdblf    = (float*)take((size_t)B_ * L_ * XD_ * 4);
  unsigned short* dtm_bf   = (unsigned short*)take((size_t)B_ * L_ * DTR_ * 2);
  unsigned short* dtow_bf  = (unsigned short*)take((size_t)DI_ * DTR_ * 2);
  float*          deltaf   = (float*)take((size_t)B_ * DI_ * L_ * 4);
  unsigned short* yT_bf    = (unsigned short*)take((size_t)B_ * L_ * DI_ * 2);
  unsigned short* outp_bf  = (unsigned short*)take((size_t)DM_ * DI_ * 2);

  // 1) bf16 conversions
  cvt_bf16<<<512, 256, 0, stream>>>(hidden,   h_bf,    B_ * L_ * DM_);
  cvt_bf16<<<512, 256, 0, stream>>>(inprojw,  inp_bf,  2 * DI_ * DM_);
  cvt_bf16<<<128, 256, 0, stream>>>(xprojw,   xp_bf,   XD_ * DI_);
  cvt_bf16<<<128, 256, 0, stream>>>(dtoutw,   dtow_bf, DI_ * DTR_);
  cvt_bf16<<<512, 256, 0, stream>>>(outprojw, outp_bf, DM_ * DI_);

  constexpr int M = B_ * L_;   // 2048

  // 2) in_proj GEMM: xz[b,e,l] = hidden @ in_proj^T  (M=2048,N=4096,K=1024)
  gemm_bf16_wmma<1, 4, 2 * DI_, DM_, L_>
      <<<(M / 16) * (2 * DI_ / 64) / WPB, WPB * 32, 0, stream>>>(h_bf, inp_bf, xzf);

  // 3) causal conv K=4 + silu; emit xc (b,d,l) fp32 and x^T (b,l,d) bf16
  conv4_silu<<<(B_ * DI_) / 256, 256, 0, stream>>>(xzf, convw, convb, xcf, xT_bf);

  // 4) x_proj GEMM: x_dbl[b,l,96] (M=2048,N=96,K=2048), NT=2 since 96 = 3*32
  gemm_bf16_wmma<0, 2, XD_, DI_, L_>
      <<<(M / 16) * (XD_ / 32) / WPB, WPB * 32, 0, stream>>>(xT_bf, xp_bf, xdblf);

  // 5) fused inner Mamba block -> dt_m bf16 (b,l,64)
  inner_mamba<<<B_, 64, 0, stream>>>(xdblf, m_inw, m_convw, m_convb, m_xprojw,
                                     m_dtw, m_dtb, m_Alog, m_Dvec, m_outw, dtm_bf);

  // 6) delta GEMM: delta[b,d,l] = dt_m @ dt_out_w^T  (M=2048,N=2048,K=64)
  gemm_bf16_wmma<1, 4, DI_, DTR_, L_>
      <<<(M / 16) * (DI_ / 64) / WPB, WPB * 32, 0, stream>>>(dtm_bf, dtow_bf, deltaf);

  // 7) outer selective scan -> y^T bf16 (b,l,d)
  scan_outer<<<B_ * (DI_ / 256), 256, 0, stream>>>(deltaf, xcf, xzf, xdblf,
                                                   Alog, Dvec, dtoutb, yT_bf);

  // 8) out_proj GEMM: out[b,l,o] = y^T @ out_proj^T  (M=2048,N=1024,K=2048)
  gemm_bf16_wmma<0, 4, DM_, DI_, L_>
      <<<(M / 16) * (DM_ / 64) / WPB, WPB * 32, 0, stream>>>(yT_bf, outp_bf, out);
}
